// Attention_25975962206449
// MI455X (gfx1250) — compile-verified, matrix operand order checked by reading, not run on hardware
//
#include <hip/hip_runtime.h>

typedef __bf16 bf16_t;
typedef __attribute__((ext_vector_type(16))) __bf16 v16bf;
typedef __attribute__((ext_vector_type(8)))  __bf16 v8bf;
typedef __attribute__((ext_vector_type(8)))  float  v8f;
typedef __attribute__((ext_vector_type(4)))  unsigned int v4u;
typedef __attribute__((ext_vector_type(8)))  int v8i;
typedef __attribute__((ext_vector_type(4)))  int v4i;

typedef __attribute__((address_space(3))) bf16_t lds_bf16_t;
typedef __attribute__((address_space(3))) void   lds_void_t;
typedef __attribute__((address_space(1))) const void glob_cvoid_t;

#define B_   8
#define N_   1024
#define C_   768
#define H_   12
#define D_   64
#define KW_  1024
#define TC_  2304          // 3*C
#define EPS_ 1e-5f

#if defined(__has_builtin)
#  if __has_builtin(__builtin_amdgcn_tensor_load_to_lds)
#    define USE_TDM 1
#  endif
#  if __has_builtin(__builtin_amdgcn_global_load_async_to_lds_b128)
#    define USE_ASYNC 1
#  endif
#endif

static __device__ __forceinline__ void wait_tensorcnt0() {
#if __has_builtin(__builtin_amdgcn_s_wait_tensorcnt)
  __builtin_amdgcn_s_wait_tensorcnt(0);
#else
  asm volatile("s_wait_tensorcnt 0x0" ::: "memory");
#endif
}

static __device__ __forceinline__ v8f wmma_bf16(v16bf a, v16bf b, v8f c) {
  // D = A(16x32 bf16) * B(32x16 bf16) + C(16x16 f32)
  return __builtin_amdgcn_wmma_f32_16x16x32_bf16(false, a, false, b, (short)0, c,
                                                 false, false);
}

// Load a 16x32 bf16 operand tile into the CDNA5 WMMA per-lane fragment layout.
// Works for A (rows contiguous over K) and B (columns contiguous over K).
// lane l: idx r = l&15 (row for A / col for B), group g = l>>4:
//   elems 0..7  = K 8g .. 8g+7
//   elems 8..15 = K 8g+16 .. 8g+23
static __device__ __forceinline__ v16bf load_frag(const bf16_t* base, int ld, int lane) {
  const int r  = lane & 15;
  const int kb = (lane >> 4) << 3;
  const bf16_t* p = base + (size_t)r * ld + kb;
  v16bf f;
#pragma unroll
  for (int i = 0; i < 8; ++i) { f[i] = p[i]; f[i + 8] = p[i + 16]; }
  return f;
}

// ---------------------------------------------------------------- converts --
__global__ void f32_to_bf16_kernel(const float* __restrict__ src,
                                   bf16_t* __restrict__ dst, int n) {
  int i = blockIdx.x * blockDim.x + threadIdx.x;
  int stride = gridDim.x * blockDim.x;
  for (; i < n; i += stride) dst[i] = (bf16_t)src[i];
}

// ------------------------------------------------------------------- QKV ----
// grid (TC_/256, (B_*N_)/16), block 256. Wave w owns 32 output cols (2 tiles).
// The 16x768 A-tile is DMA'd once per block into LDS by the Tensor Data Mover
// (TENSOR_LOAD_TO_LDS, waited with s_wait_tensorcnt), then all 8 waves read
// A-fragments from LDS while streaming B from L2.
// Writes q,k as [B,H,N,D] bf16 and v transposed as [B,H,D,N] bf16.
__global__ void qkv_gemm_kernel(const bf16_t* __restrict__ xb,
                                const bf16_t* __restrict__ wq,
                                bf16_t* __restrict__ qb,
                                bf16_t* __restrict__ kb,
                                bf16_t* __restrict__ vtb) {
  __shared__ bf16_t As[16 * C_];                  // 24 KB A staging tile
  const int lane = threadIdx.x & 31;
  const int wave = threadIdx.x >> 5;
  const int m0 = blockIdx.y * 16;                 // row in [0, B*N)
  const int n0 = blockIdx.x * 256 + wave * 32;    // col in [0, 3C)
  const bf16_t* src = xb + (size_t)m0 * C_;

#if USE_TDM
  if (wave == 0) {
    const unsigned long long ga = (unsigned long long)src;
    const unsigned int lds_off =
        (unsigned int)(unsigned long long)(lds_bf16_t*)As;
    // D# group0: count=1 | lds_addr | global_addr(57b) | type=2
    v4u g0;
    g0[0] = 1u;
    g0[1] = lds_off;
    g0[2] = (unsigned int)(ga & 0xffffffffull);
    g0[3] = (unsigned int)((ga >> 32) & 0x01ffffffull) | (2u << 30);
    // D# group1: data_size=2B; tensor 768x16; tile 768x16; dim0_stride=768
    v8i g1;
    g1[0] = 0x00010000;                           // data_size=1 (2 bytes)
    g1[1] = (int)((C_ & 0xffff) << 16);           // tensor_dim0 [15:0]
    g1[2] = (int)((C_ >> 16) | (16 << 16));       // dim0 [31:16] | tensor_dim1 lo
    g1[3] = (int)(C_ << 16);                      // tensor_dim1 hi | tile_dim0
    g1[4] = 16;                                   // tile_dim1=16, tile_dim2=0
    g1[5] = C_;                                   // tensor_dim0_stride [31:0]
    g1[6] = 0;
    g1[7] = 0;
    // degenerate higher dims (length 1, tiles unused)
    v4i g2; g2[0] = 1; g2[1] = 1; g2[2] = 0; g2[3] = 0;
    v4i g3; g3[0] = 0; g3[1] = (1 << 16); g3[2] = 0; g3[3] = 0;
#if __clang_major__ >= 23
    v8i z8 = {0, 0, 0, 0, 0, 0, 0, 0};
    __builtin_amdgcn_tensor_load_to_lds(g0, g1, g2, g3, z8, 0);
#else
    __builtin_amdgcn_tensor_load_to_lds(g0, g1, g2, g3, 0);
#endif
    wait_tensorcnt0();
  }
  __syncthreads();
#elif USE_ASYNC
  for (int i = threadIdx.x * 8; i < 16 * C_; i += 256 * 8) {
    __builtin_amdgcn_global_load_async_to_lds_b128(
        (glob_cvoid_t*)(src + i), (lds_void_t*)(As + i), 0, 0);
  }
#if __has_builtin(__builtin_amdgcn_s_wait_asynccnt)
  __builtin_amdgcn_s_wait_asynccnt(0);
#else
  asm volatile("s_wait_asynccnt 0x0" ::: "memory");
#endif
  __syncthreads();
#else
  for (int i = threadIdx.x * 8; i < 16 * C_; i += 256 * 8)
    *(v8bf*)(As + i) = *(const v8bf*)(src + i);
  __syncthreads();
#endif

  v8f acc0 = {}, acc1 = {};
  for (int k = 0; k < C_; k += 32) {
    v16bf a  = load_frag(As + k, C_, lane);       // ds reads of staged tile
    v16bf b0 = load_frag(wq + (size_t)(n0 +  0) * C_ + k, C_, lane);
    v16bf b1 = load_frag(wq + (size_t)(n0 + 16) * C_ + k, C_, lane);
    acc0 = wmma_bf16(a, b0, acc0);
    acc1 = wmma_bf16(a, b1, acc1);
  }
  const int nl = lane & 15;
  const int mb = (lane >> 4) << 3;
#pragma unroll
  for (int t = 0; t < 2; ++t) {
    v8f acc = t ? acc1 : acc0;
    const int j     = n0 + t * 16 + nl;           // col index in [0, 2304)
    const int which = j / (H_ * D_);              // 0=q 1=k 2=v  (uniform/wave)
    const int h     = (j % (H_ * D_)) / D_;
    const int dd    = j % D_;
#pragma unroll
    for (int r = 0; r < 8; ++r) {
      const int row = m0 + mb + r;                // 0..8191
      const int bb  = row >> 10, nn = row & (N_ - 1);
      const int bh  = bb * H_ + h;
      bf16_t val = (bf16_t)acc[r];
      if (which == 0)      qb [((size_t)bh * N_ + nn) * D_ + dd] = val;
      else if (which == 1) kb [((size_t)bh * N_ + nn) * D_ + dd] = val;
      else                 vtb[((size_t)bh * D_ + dd) * N_ + nn] = val;
    }
  }
}

// -------------------------------------------------- fused attention block ---
// One block per (bh, 16 query rows). 256 threads = 8 waves.
//   A: s = (q_tile . K^T) * scale             -> LDS bf16 [16][N_+8]
//   B: t = s . w_fc^T + b_fc                  -> LDS f32  [16][KW_+1]
//   C: LayerNorm(gamma,beta) + softmax (f32)  -> attn_t [B,H,KW,N] bf16
__global__ void attn_kernel(const bf16_t* __restrict__ qb,
                            const bf16_t* __restrict__ kb,
                            const bf16_t* __restrict__ wfc,
                            const float*  __restrict__ bfc,
                            const float*  __restrict__ gamma,
                            const float*  __restrict__ beta,
                            bf16_t* __restrict__ attn_t) {
  const int lane = threadIdx.x & 31;
  const int wave = threadIdx.x >> 5;
  const int bh = blockIdx.x >> 6;                 // N_/16 = 64 tiles per bh
  const int m0 = (blockIdx.x & 63) * 16;

  extern __shared__ char smem[];
  const int SLD = N_ + 8;
  const int TLD = KW_ + 1;
  bf16_t* s   = (bf16_t*)smem;                                  // 16*SLD bf16
  float*  tf  = (float*)(smem + 16 * SLD * sizeof(bf16_t));     // 16*TLD f32
  float*  red = tf + 16 * TLD;                                  // 256 f32
  float*  st  = red + 256;                                      // 32 f32

  const int nl = lane & 15;
  const int mb = (lane >> 4) << 3;
  const float scale = 0.125f;                     // d^-0.5, d = 64

  // ---- phase A: s = q.K^T * scale ------------------------------------
  const bf16_t* qtile = qb + ((size_t)bh * N_ + m0) * D_;
  v16bf a0 = load_frag(qtile + 0,  D_, lane);
  v16bf a1 = load_frag(qtile + 32, D_, lane);
#pragma unroll
  for (int t = 0; t < 8; ++t) {
    const int col0 = wave * 128 + t * 16;
    const bf16_t* kt = kb + ((size_t)bh * N_ + col0) * D_;
    v16bf b0 = load_frag(kt + 0,  D_, lane);
    v16bf b1 = load_frag(kt + 32, D_, lane);
    v8f acc = {};
    acc = wmma_bf16(a0, b0, acc);
    acc = wmma_bf16(a1, b1, acc);
    const int cc = col0 + nl;
#pragma unroll
    for (int r = 0; r < 8; ++r) s[(mb + r) * SLD + cc] = (bf16_t)(acc[r] * scale);
  }
  __syncthreads();

  // ---- phase B: t = s . w_fc^T + b_fc --------------------------------
  v8f accB[8] = {};
  for (int k = 0; k < N_; k += 32) {
    v16bf a = load_frag(s + k, SLD, lane);        // ds reads of shared s
    __builtin_prefetch(wfc + (size_t)(wave * 128) * N_ + k + 128, 0, 1);
#pragma unroll
    for (int t = 0; t < 8; ++t) {
      v16bf b = load_frag(wfc + (size_t)(wave * 128 + t * 16) * N_ + k, N_, lane);
      accB[t] = wmma_bf16(a, b, accB[t]);
    }
  }
#pragma unroll
  for (int t = 0; t < 8; ++t) {
    const int kw = wave * 128 + t * 16 + nl;
    const float bias = bfc[kw];
#pragma unroll
    for (int r = 0; r < 8; ++r) tf[(mb + r) * TLD + kw] = accB[t][r] + bias;
  }
  __syncthreads();

  // ---- phase C: LayerNorm + softmax over KW, transpose-store ---------
  const int row = threadIdx.x >> 4;               // 0..15
  const int seg = threadIdx.x & 15;               // 64 cols per thread
  float* trow = tf + row * TLD;

  float sm = 0.f, sq = 0.f;
#pragma unroll 4
  for (int i = 0; i < 64; ++i) { float v = trow[seg * 64 + i]; sm += v; sq += v * v; }
  red[row * 16 + seg] = sm;
  __syncthreads();
  if (seg == 0) {
    float tot = 0.f;
    for (int i = 0; i < 16; ++i) tot += red[row * 16 + i];
    st[row] = tot * (1.0f / KW_);                 // mean
  }
  __syncthreads();
  red[row * 16 + seg] = sq;
  __syncthreads();
  if (seg == 0) {
    float tot = 0.f;
    for (int i = 0; i < 16; ++i) tot += red[row * 16 + i];
    const float mean = st[row];
    st[16 + row] = rsqrtf(tot * (1.0f / KW_) - mean * mean + EPS_);   // rstd
  }
  __syncthreads();
  {
    const float mean = st[row], rstd = st[16 + row];
    float mx = -3.0e38f;
#pragma unroll 4
    for (int i = 0; i < 64; ++i) {
      const int c = seg * 64 + i;
      const float v = (trow[c] - mean) * rstd * gamma[c] + beta[c];
      trow[c] = v;
      mx = fmaxf(mx, v);
    }
    red[row * 16 + seg] = mx;
  }
  __syncthreads();
  if (seg == 0) {
    float m = -3.0e38f;
    for (int i = 0; i < 16; ++i) m = fmaxf(m, red[row * 16 + i]);
    st[row] = m;                                   // row max
  }
  __syncthreads();
  {
    const float rmax = st[row];
    float es = 0.f;
#pragma unroll 4
    for (int i = 0; i < 64; ++i) {
      const int c = seg * 64 + i;
      const float e = __expf(trow[c] - rmax);
      trow[c] = e;
      es += e;
    }
    red[row * 16 + seg] = es;
  }
  __syncthreads();
  if (seg == 0) {
    float tot = 0.f;
    for (int i = 0; i < 16; ++i) tot += red[row * 16 + i];
    st[16 + row] = 1.0f / tot;
  }
  __syncthreads();
  {
    const float inv = st[16 + row];
    bf16_t* outb = attn_t + (size_t)bh * KW_ * N_ + (m0 + row);
#pragma unroll 4
    for (int i = 0; i < 64; ++i) {
      const int c = seg * 64 + i;
      outb[(size_t)c * N_] = (bf16_t)(trow[c] * inv);   // transposed: [kw][n]
    }
  }
}

// ------------------------------------------------------ y = attn^T . v -----
// grid B_*H_*(KW_/32), block 256; wave -> one 16x16 tile.
// Stores y transposed as y_t [B,H,D,KW] bf16 (contiguous per lane).
__global__ void av_gemm_kernel(const bf16_t* __restrict__ attn_t,
                               const bf16_t* __restrict__ vtb,
                               bf16_t* __restrict__ yt) {
  const int lane = threadIdx.x & 31, wave = threadIdx.x >> 5;
  const int bh = blockIdx.x >> 5;
  const int m0 = (blockIdx.x & 31) * 32 + (wave >> 2) * 16;   // kw rows
  const int col0 = (wave & 3) * 16;                           // dd cols
  const bf16_t* A  = attn_t + ((size_t)bh * KW_ + m0) * N_;
  const bf16_t* Bp = vtb    + ((size_t)bh * D_ + col0) * N_;
  v8f acc = {};
  for (int k = 0; k < N_; k += 32) {
    __builtin_prefetch(A + k + 128, 0, 1);
    v16bf a = load_frag(A + k, N_, lane);
    v16bf b = load_frag(Bp + k, N_, lane);
    acc = wmma_bf16(a, b, acc);
  }
  const int dd = col0 + (lane & 15);
  const int mb = (lane >> 4) << 3;
  bf16_t* out = yt + ((size_t)bh * D_ + dd) * KW_;
#pragma unroll
  for (int r = 0; r < 8; ++r) out[m0 + mb + r] = (bf16_t)acc[r];
}

// --------------------------------- out = (w_fc2 . y) + b_fc2 -> [B,N,C] ----
// grid B_*H_*(N_/32), block 256; wave -> one 16x16 tile. f32 output.
__global__ void fc2_kernel(const bf16_t* __restrict__ wfc2,
                           const bf16_t* __restrict__ yt,
                           const float* __restrict__ bfc2,
                           float* __restrict__ out) {
  const int lane = threadIdx.x & 31, wave = threadIdx.x >> 5;
  const int bh = blockIdx.x >> 5;
  const int b = bh / H_, h = bh % H_;
  const int m0 = (blockIdx.x & 31) * 32 + (wave >> 2) * 16;   // n rows
  const int col0 = (wave & 3) * 16;                           // dd cols
  const bf16_t* A  = wfc2 + (size_t)m0 * KW_;
  const bf16_t* Bp = yt   + ((size_t)bh * D_ + col0) * KW_;
  v8f acc = {};
  for (int k = 0; k < KW_; k += 32) {
    __builtin_prefetch(A + k + 128, 0, 1);
    v16bf a = load_frag(A + k, KW_, lane);
    v16bf bb = load_frag(Bp + k, KW_, lane);
    acc = wmma_bf16(a, bb, acc);
  }
  const int dd = col0 + (lane & 15);
  const int mb = (lane >> 4) << 3;
#pragma unroll
  for (int r = 0; r < 8; ++r) {
    const int nrow = m0 + mb + r;
    out[((size_t)b * N_ + nrow) * C_ + h * D_ + dd] = acc[r] + bfc2[nrow];
  }
}

// --------------------------------------------------------------------------
extern "C" void kernel_launch(void* const* d_in, const int* in_sizes, int n_in,
                              void* d_out, int out_size, void* d_ws, size_t ws_size,
                              hipStream_t stream) {
  (void)in_sizes; (void)n_in; (void)out_size; (void)ws_size;
  const float* x     = (const float*)d_in[0];
  const float* wqkv  = (const float*)d_in[1];
  const float* wfc   = (const float*)d_in[2];
  const float* bfc   = (const float*)d_in[3];
  const float* gamma = (const float*)d_in[4];
  const float* beta  = (const float*)d_in[5];
  const float* wfc2  = (const float*)d_in[6];
  const float* bfc2  = (const float*)d_in[7];
  float* out = (float*)d_out;

  char* ws = (char*)d_ws;
  size_t off = 0;
  auto take = [&](size_t bytes) -> char* {
    char* p = ws + off;
    off += (bytes + 255) & ~(size_t)255;
    return p;
  };
  bf16_t* xb    = (bf16_t*)take((size_t)B_ * N_ * C_ * 2);
  bf16_t* wqkvb = (bf16_t*)take((size_t)TC_ * C_ * 2);
  bf16_t* wfcb  = (bf16_t*)take((size_t)KW_ * N_ * 2);
  bf16_t* wfc2b = (bf16_t*)take((size_t)N_ * KW_ * 2);
  bf16_t* qb    = (bf16_t*)take((size_t)B_ * H_ * N_ * D_ * 2);
  bf16_t* kb    = (bf16_t*)take((size_t)B_ * H_ * N_ * D_ * 2);
  bf16_t* vtb   = (bf16_t*)take((size_t)B_ * H_ * D_ * N_ * 2);
  bf16_t* attnT = (bf16_t*)take((size_t)B_ * H_ * KW_ * N_ * 2);
  bf16_t* ytb   = (bf16_t*)take((size_t)B_ * H_ * D_ * KW_ * 2);

  f32_to_bf16_kernel<<<512, 256, 0, stream>>>(x,    xb,    B_ * N_ * C_);
  f32_to_bf16_kernel<<<512, 256, 0, stream>>>(wqkv, wqkvb, TC_ * C_);
  f32_to_bf16_kernel<<<512, 256, 0, stream>>>(wfc,  wfcb,  KW_ * N_);
  f32_to_bf16_kernel<<<512, 256, 0, stream>>>(wfc2, wfc2b, N_ * KW_);

  qkv_gemm_kernel<<<dim3(TC_ / 256, (B_ * N_) / 16), 256, 0, stream>>>(
      xb, wqkvb, qb, kb, vtb);

  const size_t smem = 16 * (N_ + 8) * sizeof(bf16_t)   // s  (bf16)
                    + 16 * (KW_ + 1) * sizeof(float)   // t  (f32)
                    + 256 * sizeof(float)              // red
                    + 32 * sizeof(float);              // stats
  attn_kernel<<<B_ * H_ * (N_ / 16), 256, smem, stream>>>(
      qb, kb, wfcb, bfc, gamma, beta, attnT);

  av_gemm_kernel<<<B_ * H_ * (KW_ / 32), 256, 0, stream>>>(attnT, vtb, ytb);
  fc2_kernel<<<B_ * H_ * (N_ / 32), 256, 0, stream>>>(wfc2b, ytb, bfc2, out);
}
